// ArgmaxQuantize_21620865368350
// MI455X (gfx1250) — compile-verified, matrix operand order checked by reading, not run on hardware
//
#include <hip/hip_runtime.h>
#include <hip/hip_bf16.h>

// ---- problem constants (from reference) ----
#define BN    8
#define CDIM  256      // K of GEMM1
#define HW    4096     // 64*64
#define NDIM  1024     // codebook size
#define DDIM  256      // embedding dim
#define MTILE 128      // pixels per block
#define NCHUNK 64      // codes per chunk
#define LDA   264      // 256 + 8 halfword pad -> 528B row stride, conflict-free b128 frags

#define ZQ_ELEMS   (BN * DDIM * HW)          // 8388608
#define DIFF_OFF   ZQ_ELEMS                  // 1 element
#define IND_OFF    (ZQ_ELEMS + 1)            // 32768 elements

typedef __attribute__((ext_vector_type(16))) __bf16     v16bf;
typedef __attribute__((ext_vector_type(8)))  float      v8f;
typedef __attribute__((ext_vector_type(4)))  unsigned   u32x4;

union FragU { u32x4 u[2]; v16bf v; };

// packed fp32x2 -> bf16x2 (RNE); lowers to v_cvt_pk_bf16_f32
__device__ __forceinline__ unsigned pack_bf16(float a, float b) {
  union { __hip_bfloat162 v; unsigned u; } cv;
  cv.v = __float22bfloat162_rn(make_float2(a, b));
  return cv.u;
}

__global__ __launch_bounds__(256)
void vq_argmax_fused(const float* __restrict__ z,
                     const float* __restrict__ pw,
                     const float* __restrict__ pb,
                     const float* __restrict__ embed,
                     float* __restrict__ out) {
  __shared__ unsigned short lsA[MTILE * LDA];   // 67584 B
  __shared__ unsigned short lsB[NCHUNK * LDA];  // 33792 B
  __shared__ int            lsInd[MTILE];

  const int t    = threadIdx.x;
  const int lane = t & 31;
  const int wave = t >> 5;               // 0..7 : row tile of 16 pixels
  const int tile = blockIdx.x;           // 0..255
  const int b    = tile >> 5;
  const int hw0  = (tile & 31) * MTILE;

  const int l15     = lane & 15;
  const int halfsel = lane >> 4;         // 0 or 1
  const int kb_off  = halfsel * 8;       // WMMA 16-bit K sub-offset per lane half

  // ---------- load A tile (z) fp32 -> bf16 LDS ----------
  {
    const int m  = t & 127;              // pixel within tile (contiguous along hw)
    const int kh = t >> 7;               // 0/1
    const float* zp = z + (size_t)b * CDIM * HW + hw0 + m;
    #pragma unroll 8
    for (int i = 0; i < 64; ++i) {
      int k0 = kh * 2 + 4 * i;           // pair (k0, k0+1)
      float a0 = zp[(size_t)k0 * HW];
      float a1 = zp[(size_t)(k0 + 1) * HW];
      *(unsigned*)(&lsA[m * LDA + k0]) = pack_bf16(a0, a1);
    }
  }
  __syncthreads();

  // ---------- hoist all 8 A fragments into registers (invariant across chunks) ----------
  FragU fa[8];
  {
    const unsigned short* aRow = lsA + (wave * 16 + l15) * LDA;
    #pragma unroll
    for (int kk = 0; kk < 8; ++kk) {
      const int kb = kk * 32 + kb_off;
      fa[kk].u[0] = *(const u32x4*)(aRow + kb);
      fa[kk].u[1] = *(const u32x4*)(aRow + kb + 16);
    }
  }

  float rmax[8];
  int   ridx[8];
  #pragma unroll
  for (int r = 0; r < 8; ++r) { rmax[r] = -3.4e38f; ridx[r] = 0; }

  // per-thread B-chunk mapping: 16 rows x one float4 along k
  const int k4 = (t & 63) * 4;
  const int r0 = t >> 6;                 // 0..3

  // ---------- prefetch + convert chunk 0 into registers ----------
  uint2 pfp[16];
  #pragma unroll 4
  for (int i = 0; i < 16; ++i) {
    const float4 f = *(const float4*)(pw + (size_t)(r0 + 4 * i) * CDIM + k4);
    pfp[i].x = pack_bf16(f.x, f.y);
    pfp[i].y = pack_bf16(f.z, f.w);
  }

  const unsigned short* bRowBase = lsB + l15 * LDA;

  // ---------- 16 chunks of 64 codes ----------
  for (int nc = 0; nc < 16; ++nc) {
    if (nc) __syncthreads();             // prior chunk's fragment reads done
    #pragma unroll 4
    for (int i = 0; i < 16; ++i)
      *(uint2*)(&lsB[(r0 + 4 * i) * LDA + k4]) = pfp[i];
    __syncthreads();

    // prefetch + convert next chunk while WMMAs run (cvt co-executes with WMMA)
    if (nc < 15) {
      const float* pwn = pw + (size_t)(nc + 1) * NCHUNK * CDIM;
      #pragma unroll 4
      for (int i = 0; i < 16; ++i) {
        const float4 f = *(const float4*)(pwn + (size_t)(r0 + 4 * i) * CDIM + k4);
        pfp[i].x = pack_bf16(f.x, f.y);
        pfp[i].y = pack_bf16(f.z, f.w);
      }
    }

    v8f acc0 = {0,0,0,0,0,0,0,0};
    v8f acc1 = {0,0,0,0,0,0,0,0};
    v8f acc2 = {0,0,0,0,0,0,0,0};
    v8f acc3 = {0,0,0,0,0,0,0,0};

    #pragma unroll
    for (int kk = 0; kk < 8; ++kk) {
      const int kb = kk * 32 + kb_off;
      const unsigned short* bR = bRowBase + kb;
      FragU fb0, fb1, fb2, fb3;
      fb0.u[0] = *(const u32x4*)(bR);                fb0.u[1] = *(const u32x4*)(bR + 16);
      fb1.u[0] = *(const u32x4*)(bR + 16 * LDA);     fb1.u[1] = *(const u32x4*)(bR + 16 * LDA + 16);
      fb2.u[0] = *(const u32x4*)(bR + 32 * LDA);     fb2.u[1] = *(const u32x4*)(bR + 32 * LDA + 16);
      fb3.u[0] = *(const u32x4*)(bR + 48 * LDA);     fb3.u[1] = *(const u32x4*)(bR + 48 * LDA + 16);

      acc0 = __builtin_amdgcn_wmma_f32_16x16x32_bf16(false, fa[kk].v, false, fb0.v, (short)0, acc0, false, false);
      acc1 = __builtin_amdgcn_wmma_f32_16x16x32_bf16(false, fa[kk].v, false, fb1.v, (short)0, acc1, false, false);
      acc2 = __builtin_amdgcn_wmma_f32_16x16x32_bf16(false, fa[kk].v, false, fb2.v, (short)0, acc2, false, false);
      acc3 = __builtin_amdgcn_wmma_f32_16x16x32_bf16(false, fa[kk].v, false, fb3.v, (short)0, acc3, false, false);
    }

    // running per-lane argmax (n ascending within lane => jnp.argmax first-max semantics)
    const int nbase = nc * 64 + l15;
    const float b0 = pb[nbase];
    const float b1 = pb[nbase + 16];
    const float b2 = pb[nbase + 32];
    const float b3 = pb[nbase + 48];
    #pragma unroll
    for (int r = 0; r < 8; ++r) {
      float v0 = acc0[r] + b0; if (v0 > rmax[r]) { rmax[r] = v0; ridx[r] = nbase;      }
      float v1 = acc1[r] + b1; if (v1 > rmax[r]) { rmax[r] = v1; ridx[r] = nbase + 16; }
      float v2 = acc2[r] + b2; if (v2 > rmax[r]) { rmax[r] = v2; ridx[r] = nbase + 32; }
      float v3 = acc3[r] + b3; if (v3 > rmax[r]) { rmax[r] = v3; ridx[r] = nbase + 48; }
    }
  }

  // ---------- cross-lane (max, idx) reduction within 16-lane halves ----------
  #pragma unroll
  for (int r = 0; r < 8; ++r) {
    float m = rmax[r];
    int   i = ridx[r];
    #pragma unroll
    for (int off = 1; off <= 8; off <<= 1) {
      float om = __shfl_xor(m, off, 32);
      int   oi = __shfl_xor(i, off, 32);
      if (om > m || (om == m && oi < i)) { m = om; i = oi; }
    }
    if (l15 == 0) lsInd[wave * 16 + halfsel * 8 + r] = i;  // half0 -> rows 0..7, half1 -> 8..15
  }
  __syncthreads();

  // ---------- gather embed[ind] -> z_q, emit ind and diff ----------
  {
    const int i  = t & 127;              // pixel in tile (lanes contiguous along hw)
    const int dh = t >> 7;               // d in [dh*128, dh*128+128)
    const int idx = lsInd[i];
    const float* erow = embed + (size_t)idx * DDIM + dh * 128;
    float* op = out + (size_t)b * DDIM * HW + (size_t)dh * 128 * HW + hw0 + i;
    #pragma unroll 8
    for (int d4 = 0; d4 < 128; d4 += 4) {
      const float4 e = *(const float4*)(erow + d4);
      op[(size_t)(d4 + 0) * HW] = e.x;
      op[(size_t)(d4 + 1) * HW] = e.y;
      op[(size_t)(d4 + 2) * HW] = e.z;
      op[(size_t)(d4 + 3) * HW] = e.w;
    }
    if (t < MTILE) out[(size_t)IND_OFF + (size_t)tile * MTILE + t] = (float)lsInd[t];
    if (tile == 0 && t == 0) out[DIFF_OFF] = 0.0f;
  }
}

extern "C" void kernel_launch(void* const* d_in, const int* in_sizes, int n_in,
                              void* d_out, int out_size, void* d_ws, size_t ws_size,
                              hipStream_t stream) {
  const float* z     = (const float*)d_in[0];
  const float* pw    = (const float*)d_in[1];
  const float* pb    = (const float*)d_in[2];
  const float* embed = (const float*)d_in[3];
  float* out = (float*)d_out;

  dim3 grid(256);   // 32768 pixels / 128 per block
  dim3 block(256);  // 8 waves
  vq_argmax_fused<<<grid, block, 0, stream>>>(z, pw, pb, embed, out);
}
